// TemporalAttention_63694365000009
// MI455X (gfx1250) — matrix operand order repaired and checked by reading, not compiled, not run on hardware
//
#include <hip/hip_runtime.h>
#include <cstdint>

// ---------- types ----------
typedef __attribute__((ext_vector_type(16))) __bf16 v16bf;
typedef __attribute__((ext_vector_type(8)))  __bf16 v8bf;
typedef __attribute__((ext_vector_type(8)))  float  v8f;
typedef __attribute__((ext_vector_type(4)))  float  v4f;

#define NN 8192
#define DD 1024
#define TD_INV   (1.0f / 86400.0f)
#define SCALE_INV 0.03125f   // 1/sqrt(1024)
#define QS_LD 1032           // padded LDS row stride (bf16 elems): 2064B -> conflict-free

// ---------- WMMA wrapper ----------
__device__ __forceinline__ v8f wmma_bf16(v16bf a, v16bf b, v8f c) {
  return __builtin_amdgcn_wmma_f32_16x16x32_bf16(false, a, false, b, (short)0, c,
                                                 false, false);
}

// ---------- CDNA5 async global->LDS copy (ASYNCcnt path) ----------
__device__ __forceinline__ void async_copy_b128(void* lds_dst, const void* gsrc) {
  const unsigned loff = (unsigned)(uintptr_t)lds_dst;           // low 32b = LDS offset
  const unsigned long long ga = (unsigned long long)(uintptr_t)gsrc;
  asm volatile("global_load_async_to_lds_b128 %0, %1, off"
               :: "v"(loff), "v"(ga) : "memory");
}
__device__ __forceinline__ void wait_async0() {
  asm volatile("s_wait_asynccnt 0x0" ::: "memory");
}

// ---------- fragment loaders ----------
// A (16x32 bf16): lane<16: M=lane, K=[k0..k0+15]
//                 lane>=16: M=lane-16, K=[k0+16..k0+31]
__device__ __forceinline__ v16bf load_a_bf16(const __bf16* base, int ld, int row,
                                             int k0, int lane) {
  const int half = lane >> 4;
  const __bf16* p = base + (size_t)row * ld + k0 + half * 16;
  v8bf lo = *(const v8bf*)(p);
  v8bf hi = *(const v8bf*)(p + 8);
  v16bf a;
#pragma unroll
  for (int i = 0; i < 8; ++i) { a[i] = lo[i]; a[8 + i] = hi[i]; }
  return a;
}

// B (32x16 bf16): column N = lane%16 = row (col0+lane%16) of row-major source;
// lane<16 holds K=k0..k0+15, lane>=16 holds K=k0+16..k0+31 (one 32B load).
__device__ __forceinline__ v16bf load_b_bf16(const __bf16* base, int ld, int col0,
                                             int k0, int lane) {
  const int half = lane >> 4;
  const int c = lane & 15;
  const __bf16* p = base + (size_t)(col0 + c) * ld + k0 + half * 16;
  return *(const v16bf*)(p);
}

// ============================================================================
// Kernel 0: fp32 -> bf16 conversion (x and the three weight matrices)
// ============================================================================
__global__ __launch_bounds__(256) void cvt_bf16(const float* __restrict__ src,
                                                __bf16* __restrict__ dst, int n) {
  const int i = (blockIdx.x * 256 + threadIdx.x) * 8;
  if (i >= n) return;
  v4f a0 = *(const v4f*)(src + i);
  v4f a1 = *(const v4f*)(src + i + 4);
  v8bf o;
#pragma unroll
  for (int j = 0; j < 4; ++j) { o[j] = (__bf16)a0[j]; o[4 + j] = (__bf16)a1[j]; }
  *(v8bf*)(dst + i) = o;
}

// ============================================================================
// Kernel 1: QKV projection on bf16 operands.  y = x @ W^T + b
// grid = (64, 16, 3), block = 256 (8 waves). V written transposed (bf16).
// ============================================================================
__global__ __launch_bounds__(256) void proj_qkv(
    const __bf16* __restrict__ xb,
    const __bf16* __restrict__ Wqb, const __bf16* __restrict__ Wkb,
    const __bf16* __restrict__ Wvb,
    const float* __restrict__ bq, const float* __restrict__ bk,
    const float* __restrict__ bv,
    __bf16* __restrict__ Qb, __bf16* __restrict__ Kb, __bf16* __restrict__ VTb) {
  const int lane = threadIdx.x & 31;
  const int wave = threadIdx.x >> 5;
  const int half = lane >> 4;
  const int c = lane & 15;
  const int z = blockIdx.z;
  const __bf16* W = (z == 0) ? Wqb : (z == 1) ? Wkb : Wvb;
  const float* bias = (z == 0) ? bq : (z == 1) ? bk : bv;

  const int m0 = blockIdx.x * 128 + wave * 16;
  const int j0 = blockIdx.y * 64;

  v8f acc[4];
#pragma unroll
  for (int t = 0; t < 4; ++t)
#pragma unroll
    for (int i = 0; i < 8; ++i) acc[t][i] = 0.0f;

  for (int d0 = 0; d0 < DD; d0 += 32) {
    v16bf a = load_a_bf16(xb, DD, m0 + c, d0, lane);
#pragma unroll
    for (int t = 0; t < 4; ++t) {
      v16bf b = load_b_bf16(W, DD, j0 + 16 * t, d0, lane);
      acc[t] = wmma_bf16(a, b, acc[t]);
    }
  }

#pragma unroll
  for (int t = 0; t < 4; ++t) {
    const int j = j0 + 16 * t + c;
    const float bj = bias[j];
#pragma unroll
    for (int r = 0; r < 8; ++r) {
      const int m = m0 + r + 8 * half;
      const float y = acc[t][r] + bj;
      if (z == 0)      Qb[(size_t)m * DD + j] = (__bf16)y;
      else if (z == 1) Kb[(size_t)m * DD + j] = (__bf16)y;
      else             VTb[(size_t)j * NN + m] = (__bf16)y;
    }
  }
}

// ---------- cooperative async staging of a 16-row Q tile into LDS ----------
__device__ __forceinline__ void stage_q_tile(__bf16 (*Qs)[QS_LD],
                                             const __bf16* Qb, int m0, int tid) {
  const int row = tid >> 4;        // 0..15
  const int seg = tid & 15;        // 16 segments of 128B per 2KB row
  const __bf16* g = Qb + (size_t)(m0 + row) * DD + seg * 64;
  __bf16* l = &Qs[row][seg * 64];
#pragma unroll
  for (int i = 0; i < 8; ++i) async_copy_b128(l + i * 8, g + i * 8);
  wait_async0();                   // each wave drains its own ASYNCcnt
  __syncthreads();                 // then all waves agree the tile is ready
}

// ============================================================================
// Kernel 2: per-row softmax stats. grid = 512, block = 256 (8 waves).
// WG owns 16 rows (Q tile staged in LDS); wave w scans columns
// [w*1024, (w+1)*1024); per-lane online (max,sum); 16-lane shfl merge;
// cross-wave merge through LDS.  bias log(exp(-|dt|/tau)+eps) ~= -|dt|/tau.
// ============================================================================
__global__ __launch_bounds__(256) void attn_stats(
    const __bf16* __restrict__ Qb, const __bf16* __restrict__ Kb,
    const float* __restrict__ ts,
    float* __restrict__ rmax, float* __restrict__ rsum) {
  __shared__ __bf16 Qs[16][QS_LD];
  __shared__ float Sm[8][16];
  __shared__ float Ss[8][16];

  const int tid = threadIdx.x;
  const int lane = tid & 31;
  const int wave = tid >> 5;
  const int half = lane >> 4;
  const int c = lane & 15;
  const int m0 = blockIdx.x * 16;

  stage_q_tile(Qs, Qb, m0, tid);

  float tr[8], mx[8], sm[8];
#pragma unroll
  for (int r = 0; r < 8; ++r) {
    tr[r] = ts[m0 + r + 8 * half];
    mx[r] = -3.0e38f;
    sm[r] = 0.0f;
  }

  const int nlo = wave * (NN / 8);
  for (int n0 = nlo; n0 < nlo + NN / 8; n0 += 64) {
    v8f acc[4];
#pragma unroll
    for (int t = 0; t < 4; ++t)
#pragma unroll
      for (int i = 0; i < 8; ++i) acc[t][i] = 0.0f;

    for (int d0 = 0; d0 < DD; d0 += 32) {
      v16bf a = load_a_bf16(&Qs[0][0], QS_LD, c, d0, lane);   // from LDS
#pragma unroll
      for (int t = 0; t < 4; ++t) {
        v16bf b = load_b_bf16(Kb, DD, n0 + 16 * t, d0, lane);
        acc[t] = wmma_bf16(a, b, acc[t]);
      }
    }

#pragma unroll
    for (int t = 0; t < 4; ++t) {
      const float tc = ts[n0 + 16 * t + c];
#pragma unroll
      for (int r = 0; r < 8; ++r) {
        const float s = acc[t][r] * SCALE_INV - fabsf(tr[r] - tc) * TD_INV;
        const float nm = fmaxf(mx[r], s);
        sm[r] = sm[r] * __expf(mx[r] - nm) + __expf(s - nm);
        mx[r] = nm;
      }
    }
  }

  // merge the 16 lanes sharing each row (xor masks < 16 stay within the half)
#pragma unroll
  for (int off = 1; off < 16; off <<= 1) {
#pragma unroll
    for (int r = 0; r < 8; ++r) {
      const float om = __shfl_xor(mx[r], off, 32);
      const float os = __shfl_xor(sm[r], off, 32);
      const float nm = fmaxf(mx[r], om);
      sm[r] = sm[r] * __expf(mx[r] - nm) + os * __expf(om - nm);
      mx[r] = nm;
    }
  }
  if (c == 0) {
#pragma unroll
    for (int r = 0; r < 8; ++r) {
      Sm[wave][r + 8 * half] = mx[r];
      Ss[wave][r + 8 * half] = sm[r];
    }
  }
  __syncthreads();

  // cross-wave merge: 16 threads, one per row
  if (tid < 16) {
    float m = -3.0e38f, s = 0.0f;
#pragma unroll
    for (int w = 0; w < 8; ++w) {
      const float om = Sm[w][tid];
      const float os = Ss[w][tid];
      const float nm = fmaxf(m, om);
      s = s * __expf(m - nm) + os * __expf(om - nm);
      m = nm;
    }
    rmax[m0 + tid] = m;
    rsum[m0 + tid] = s;
  }
}

// ============================================================================
// Kernel 3: out = softmax(S) @ V.  grid = 512, block = 256 (8 waves).
// Q tile staged in LDS (async); per 256-col chunk each wave computes a 16x32
// S slab -> P(bf16) in LDS; all waves then run P @ V^T WMMAs into resident
// fp32 accumulators (16 rows x 128 cols per wave).
// ============================================================================
__global__ __launch_bounds__(256) void attn_out(
    const __bf16* __restrict__ Qb, const __bf16* __restrict__ Kb,
    const __bf16* __restrict__ VTb, const float* __restrict__ ts,
    const float* __restrict__ rmax, const float* __restrict__ rsum,
    float* __restrict__ out) {
  __shared__ __bf16 Qs[16][QS_LD];
  __shared__ __bf16 Plds[16][272];

  const int tid = threadIdx.x;
  const int lane = tid & 31;
  const int wave = tid >> 5;
  const int half = lane >> 4;
  const int c = lane & 15;
  const int m0 = blockIdx.x * 16;
  const int j0 = wave * 128;

  stage_q_tile(Qs, Qb, m0, tid);

  float tr[8], rm[8], ri[8];
#pragma unroll
  for (int r = 0; r < 8; ++r) {
    const int row = m0 + r + 8 * half;
    tr[r] = ts[row];
    rm[r] = rmax[row];
    ri[r] = 1.0f / rsum[row];
  }

  v8f acc[8];
#pragma unroll
  for (int t = 0; t < 8; ++t)
#pragma unroll
    for (int i = 0; i < 8; ++i) acc[t][i] = 0.0f;

  for (int nc = 0; nc < NN; nc += 256) {
    // ---- S slab: this wave's 32 columns ----
    const int nbase = nc + wave * 32;
    v8f sa[2];
#pragma unroll
    for (int t = 0; t < 2; ++t)
#pragma unroll
      for (int i = 0; i < 8; ++i) sa[t][i] = 0.0f;

    for (int d0 = 0; d0 < DD; d0 += 32) {
      v16bf a = load_a_bf16(&Qs[0][0], QS_LD, c, d0, lane);   // from LDS
#pragma unroll
      for (int t = 0; t < 2; ++t) {
        v16bf b = load_b_bf16(Kb, DD, nbase + 16 * t, d0, lane);
        sa[t] = wmma_bf16(a, b, sa[t]);
      }
    }

#pragma unroll
    for (int t = 0; t < 2; ++t) {
      const float tc = ts[nbase + 16 * t + c];
#pragma unroll
      for (int r = 0; r < 8; ++r) {
        const float s = sa[t][r] * SCALE_INV - fabsf(tr[r] - tc) * TD_INV;
        const float p = __expf(s - rm[r]) * ri[r];
        Plds[r + 8 * half][wave * 32 + 16 * t + c] = (__bf16)p;
      }
    }
    __syncthreads();

    // ---- P @ V over the 256-wide chunk ----
    const __bf16* pbase = &Plds[0][0];
    for (int k0 = 0; k0 < 256; k0 += 32) {
      v16bf a = load_a_bf16(pbase, 272, c, k0, lane);
#pragma unroll
      for (int t = 0; t < 8; ++t) {
        v16bf b = load_b_bf16(VTb, NN, j0 + 16 * t, nc + k0, lane);
        acc[t] = wmma_bf16(a, b, acc[t]);
      }
    }
    __syncthreads();
  }

#pragma unroll
  for (int t = 0; t < 8; ++t)
#pragma unroll
    for (int r = 0; r < 8; ++r)
      out[(size_t)(m0 + r + 8 * half) * DD + j0 + 16 * t + c] = acc[t][r];
}

// ============================================================================
extern "C" void kernel_launch(void* const* d_in, const int* in_sizes, int n_in,
                              void* d_out, int out_size, void* d_ws, size_t ws_size,
                              hipStream_t stream) {
  (void)in_sizes; (void)n_in; (void)out_size; (void)ws_size;
  const float* x  = (const float*)d_in[0];
  const float* ts = (const float*)d_in[1];
  const float* Wq = (const float*)d_in[2];
  const float* bq = (const float*)d_in[3];
  const float* Wk = (const float*)d_in[4];
  const float* bk = (const float*)d_in[5];
  const float* Wv = (const float*)d_in[6];
  const float* bv = (const float*)d_in[7];
  float* out = (float*)d_out;

  char* ws = (char*)d_ws;
  const size_t mat = (size_t)NN * DD * sizeof(__bf16);   // 16 MB
  const size_t wsz = (size_t)DD * DD * sizeof(__bf16);   // 2 MB
  __bf16* Qb  = (__bf16*)(ws);
  __bf16* Kb  = (__bf16*)(ws + mat);
  __bf16* VTb = (__bf16*)(ws + 2 * mat);
  __bf16* xb  = (__bf16*)(ws + 3 * mat);
  __bf16* Wqb = (__bf16*)(ws + 4 * mat);
  __bf16* Wkb = (__bf16*)(ws + 4 * mat + wsz);
  __bf16* Wvb = (__bf16*)(ws + 4 * mat + 2 * wsz);
  float* rmax = (float*)(ws + 4 * mat + 3 * wsz);
  float* rsum = rmax + NN;

  // fp32 -> bf16 staging
  cvt_bf16<<<(NN * DD) / 2048, 256, 0, stream>>>(x, xb, NN * DD);
  cvt_bf16<<<(DD * DD) / 2048, 256, 0, stream>>>(Wq, Wqb, DD * DD);
  cvt_bf16<<<(DD * DD) / 2048, 256, 0, stream>>>(Wk, Wkb, DD * DD);
  cvt_bf16<<<(DD * DD) / 2048, 256, 0, stream>>>(Wv, Wvb, DD * DD);

  proj_qkv<<<dim3(NN / 128, DD / 64, 3), 256, 0, stream>>>(
      xb, Wqb, Wkb, Wvb, bq, bk, bv, Qb, Kb, VTb);
  attn_stats<<<NN / 16, 256, 0, stream>>>(Qb, Kb, ts, rmax, rsum);
  attn_out<<<NN / 16, 256, 0, stream>>>(Qb, Kb, VTb, ts, rmax, rsum, out);
}
